// BiCameralCRSN_24902220382469
// MI455X (gfx1250) — matrix operand order repaired and checked by reading, not compiled, not run on hardware
//
#include <hip/hip_runtime.h>
#include <hip/hip_bf16.h>
#include <math.h>

// ---------------------------------------------------------------------------
// BiCameralCRSN for MI455X (gfx1250, wave32, WMMA).
//   Phase 1: pack_weights       - f32 -> f16, N-major repack (WMMA B-frag friendly)
//   Phase 2: crsn_recurrent     - 1 persistent block, 8 waves; all D-sized GEMMs
//                                 via v_wmma_f32_16x16x32_f16; everything else VALU f32
//   Phase 3: crsn_decoder       - (S*B x 2D) @ (2D x V) WMMA GEMM, store-bound
// Workspace layout (bytes), assumes ws_size >= ~13 MB:
//   [0)        packed f16 weights: 6*16384 + 65536 + 4194304 halfs
//   [8716288)  zq f16: 128*32*256 halfs
//   [10813440) f32: mem(393216) T(8192) U(8192) ACC(8192) SL(8192) G(4096)
// ---------------------------------------------------------------------------

typedef __attribute__((ext_vector_type(16))) _Float16 v16h;
typedef __attribute__((ext_vector_type(8)))  _Float16 v8h;
typedef __attribute__((ext_vector_type(8)))  float    v8f;

union AFrag { v16h v; v8h h2[2]; _Float16 h[16]; };

static __device__ inline v8f wmma16(v16h a, v16h b, v8f c) {
  // D = A(16x32 f16) * B(32x16 f16) + C(16x16 f32)
  return __builtin_amdgcn_wmma_f32_16x16x32_f16(
      /*neg_a=*/false, a, /*neg_b=*/false, b,
      /*c_mod=*/(short)0, c, /*reuse_a=*/false, /*reuse_b=*/false);
}

// A fragment from row-major f16 buffer (LDS): rows 0..31, row stride LD halfs.
// A 16x32 frag, contiguous K order chosen here: lanes 0-15 = M, K contiguous run {sel*16..+15} (matches B).
static __device__ inline v16h load_afrag(const _Float16* X, int LD, int co,
                                         int mt, int kb, int lane) {
  int row = mt * 16 + (lane & 15);
  int sel = lane >> 4;
  const _Float16* p = X + row * LD + co + kb * 32 + sel * 16;
  AFrag a;
  a.h2[0] = *(const v8h*)p;
  a.h2[1] = *(const v8h*)(p + 8);
  return a.v;
}

// B fragment from packed N-major weights Wt[n*K + k] (global f16).
// B 32x16 frag, contiguous K order chosen here (same as A): lane = column n; lanes<16 hold K kb*32..+15,
// lanes>=16 hold K kb*32+16..+31 -> one contiguous 16-half (32B) run per lane.
static __device__ inline v16h load_bfrag(const _Float16* Wt, int K,
                                         int colBase, int kb, int lane) {
  int col = colBase + (lane & 15);
  int k0  = kb * 32 + (lane >> 4) * 16;
  const _Float16* p = Wt + (size_t)col * K + k0;
  AFrag a;
  a.h2[0] = *(const v8h*)p;
  a.h2[1] = *(const v8h*)(p + 8);
  return a.v;
}

#define GFENCE() do { __threadfence(); __syncthreads(); } while (0)

// Complex linear: out_r = r@Wr - i@Wi ; out_i = i@Wr + r@Wi.
// M=32, N=128, K=128. A in LDS (stride 512, col offsets coR/coI).
// Outputs to global f32, row stride 256 (caller bakes column offset).
static __device__ void clin_gemm(const _Float16* X, int coR, int coI,
                                 const _Float16* WtR, const _Float16* WtI,
                                 float* outR, float* outI) {
  int tid = threadIdx.x, lane = tid & 31, wave = tid >> 5;
  for (int tile = wave; tile < 16; tile += 8) {
    int mt = tile & 1, nt = tile >> 1;
    v8f aRR = {}, aII = {}, aIR = {}, aRI = {};
    for (int kb = 0; kb < 4; ++kb) {
      v16h Ar = load_afrag(X, 512, coR, mt, kb, lane);
      v16h Ai = load_afrag(X, 512, coI, mt, kb, lane);
      v16h Br = load_bfrag(WtR, 128, nt * 16, kb, lane);
      v16h Bi = load_bfrag(WtI, 128, nt * 16, kb, lane);
      aRR = wmma16(Ar, Br, aRR);
      aII = wmma16(Ai, Bi, aII);
      aIR = wmma16(Ai, Br, aIR);
      aRI = wmma16(Ar, Bi, aRI);
    }
    int col = nt * 16 + (lane & 15);
    int rbase = mt * 16 + ((lane >> 4) ? 8 : 0);
    for (int r = 0; r < 8; ++r) {
      outR[(rbase + r) * 256 + col] = aRR[r] - aII[r];   // f16 WMMA has no A/B NEG
      outI[(rbase + r) * 256 + col] = aIR[r] + aRI[r];   // -> combine in VALU
    }
  }
}

// Plain GEMM: M=32 x K x N, A in LDS (stride LD), packed Wt, out f32 stride ldo.
static __device__ void gemm_nk(const _Float16* X, int LD, const _Float16* Wt,
                               int K, int N, float* out, int ldo) {
  int tid = threadIdx.x, lane = tid & 31, wave = tid >> 5;
  int tiles = 2 * (N / 16);
  for (int tile = wave; tile < tiles; tile += 8) {
    int mt = tile & 1, nt = tile >> 1;
    v8f acc = {};
    for (int kb = 0; kb < K / 32; ++kb) {
      v16h A = load_afrag(X, LD, 0, mt, kb, lane);
      v16h B = load_bfrag(Wt, K, nt * 16, kb, lane);
      acc = wmma16(A, B, acc);
    }
    int col = nt * 16 + (lane & 15);
    int rbase = mt * 16 + ((lane >> 4) ? 8 : 0);
    for (int r = 0; r < 8; ++r) out[(rbase + r) * ldo + col] = acc[r];
  }
}

// ---------------------------------------------------------------------------
// Phase 1: pack weights to f16, transposed to N-major (Wt[n*K+k] = W[k*N+n]).
// segs: 6x cell (128x128), gate (512x128), dec (256x16384)
// ---------------------------------------------------------------------------
__global__ void pack_weights(const float* cWr, const float* cWi,
                             const float* vWr, const float* vWi,
                             const float* oWr, const float* oWi,
                             const float* gateW, const float* decW,
                             _Float16* pk) {
  const float* cell[6] = {cWr, cWi, vWr, vWi, oWr, oWi};
  _Float16* gateWt = pk + 98304;
  _Float16* decWt  = pk + 163840;
  for (size_t idx = (size_t)blockIdx.x * 256 + threadIdx.x; idx < 4358144ull;
       idx += (size_t)gridDim.x * 256) {
    if (idx < 98304) {
      int seg = (int)(idx >> 14), l = (int)(idx & 16383);
      int nn = l >> 7, kk = l & 127;
      pk[idx] = (_Float16)cell[seg][kk * 128 + nn];
    } else if (idx < 163840) {
      int l = (int)(idx - 98304);
      int nn = l >> 9, kk = l & 511;
      gateWt[l] = (_Float16)gateW[kk * 128 + nn];
    } else {
      size_t l = idx - 163840;
      int nn = (int)(l >> 8), kk = (int)(l & 255);
      decWt[l] = (_Float16)decW[(size_t)kk * 16384 + nn];
    }
  }
}

// ---------------------------------------------------------------------------
// VQ: d = |zf|^2 + |cb|^2 - 2 zf.cb - GB*sigmoid(adj[prev]) - CG*softmax(zf@cW+b)
// straight-through output == cb[argmin]; loss = (1+CC)*mean((zq-zf)^2)
// ---------------------------------------------------------------------------
static __device__ void vq_step(const float* zf, const float* cb, const float* cW,
                               const float* cbias, const float* adj, int NC,
                               const int* prevIdx, int* newIdx,
                               float* Sc, float* Fx, float* Red, float* RowZn,
                               float* lossAcc) {
  int tid = threadIdx.x;
  if (tid < 32) {
    float s = 0.f;
    for (int k = 0; k < 256; ++k) { float v = zf[tid * 256 + k]; s += v * v; }
    RowZn[tid] = s;
  }
  __syncthreads();
  for (int idx = tid; idx < 32 * NC; idx += 256) {
    int b = idx / NC, j = idx % NC;
    float sc = cbias[j], dot = 0.f, cn = 0.f;
    for (int k = 0; k < 256; ++k) {
      float z = zf[b * 256 + k], c = cb[j * 256 + k];
      sc += z * cW[k * NC + j];
      dot += z * c; cn += c * c;
    }
    Sc[b * NC + j] = sc;
    Fx[b * NC + j] = RowZn[b] + cn - 2.f * dot;
  }
  __syncthreads();
  if (tid < 32) {
    float mx = -1e30f;
    for (int j = 0; j < NC; ++j) mx = fmaxf(mx, Sc[tid * NC + j]);
    float s = 0.f;
    for (int j = 0; j < NC; ++j) { float e = expf(Sc[tid * NC + j] - mx); Sc[tid * NC + j] = e; s += e; }
    float inv = 1.f / s;
    for (int j = 0; j < NC; ++j) Sc[tid * NC + j] *= inv;
  }
  __syncthreads();
  for (int idx = tid; idx < 32 * NC; idx += 256) {
    int b = idx / NC, j = idx % NC;
    float a = adj[prevIdx[b] * NC + j];
    Fx[idx] -= 0.8f * (1.f / (1.f + expf(-a))) + 2.0f * Sc[idx];
  }
  __syncthreads();
  if (tid < 32) {
    int bi = 0; float bd = Fx[tid * NC];
    for (int j = 1; j < NC; ++j) { float d = Fx[tid * NC + j]; if (d < bd) { bd = d; bi = j; } }
    newIdx[tid] = bi;
  }
  __syncthreads();
  float p = 0.f;
  for (int e = tid; e < 8192; e += 256) {
    int b = e >> 8;
    float q = cb[newIdx[b] * 256 + (e & 255)] - zf[e];
    p += q * q;
  }
  Red[tid] = p;
  __syncthreads();
  if (tid == 0) {
    float s = 0.f;
    for (int i = 0; i < 256; ++i) s += Red[i];
    *lossAcc += 1.25f * s / 8192.f;      // (1+CC) * mean
  }
  __syncthreads();
}

// ---------------------------------------------------------------------------
// Phase 2: persistent recurrent kernel. One block, 256 threads = 8 wave32.
// ---------------------------------------------------------------------------
__global__ __launch_bounds__(256, 1) void crsn_recurrent(
    const int* __restrict__ input_ids, const int* __restrict__ prev_syn,
    const int* __restrict__ prev_sem,
    const float* __restrict__ emb_mag, const float* __restrict__ emb_phase,
    const float* __restrict__ norm_scale, const float* __restrict__ norm_shift,
    const float* __restrict__ modrelu_b,
    const float* __restrict__ halt_W, const float* __restrict__ halt_b,
    const float* __restrict__ stk_W, const float* __restrict__ stk_b,
    const float* __restrict__ cb_syn, const float* __restrict__ cb_sem,
    const float* __restrict__ ctxsyn_W, const float* __restrict__ ctxsyn_b,
    const float* __restrict__ ctxsem_W, const float* __restrict__ ctxsem_b,
    const float* __restrict__ adj_syn, const float* __restrict__ adj_sem,
    const float* __restrict__ gate_b,
    const _Float16* __restrict__ cWr_t, const _Float16* __restrict__ cWi_t,
    const _Float16* __restrict__ vWr_t, const _Float16* __restrict__ vWi_t,
    const _Float16* __restrict__ oWr_t, const _Float16* __restrict__ oWi_t,
    const _Float16* __restrict__ gateWt,
    _Float16* __restrict__ zq16, float* __restrict__ memW,
    float* __restrict__ T, float* __restrict__ U,
    float* __restrict__ ACC, float* __restrict__ SL,
    float* __restrict__ G, float* __restrict__ lossOut) {

  __shared__ __align__(16) _Float16 Xh[32 * 512];  // activations (f16 GEMM A)
  __shared__ float Sc[32 * 128];                   // multi-use scratch
  __shared__ float Ptr[32 * 48];
  __shared__ float Red[256], Red2[256];
  __shared__ float RowMean[32], RowRstd[32], RowHalt[32];
  __shared__ float RowPush[32], RowPop[32], RowNoop[32];
  __shared__ float RowRem[32], RowSum[32], RowZn[32];
  __shared__ int   PrevSyn[32], PrevSem[32], NewSyn[32], NewSem[32], Ids[32];
  __shared__ float LossAcc;

  float* Fx = (float*)Xh;   // VQ distance scratch (aliases dead f16 acts)
  int tid = threadIdx.x;

  if (tid < 32) { PrevSyn[tid] = prev_syn[tid]; PrevSem[tid] = prev_sem[tid]; }
  if (tid == 0) LossAcc = 0.f;
  for (int e = tid; e < 1536; e += 256) Ptr[e] = ((e % 48) == 0) ? 1.f : 0.f;
  for (int e = tid; e < 393216; e += 256) memW[e] = 0.f;
  GFENCE();

  for (int t = 0; t < 128; ++t) {
    // ---- timestep init + embedding: z = mag * (cos(ph), sin(ph)) ----
    if (tid < 32) { RowRem[tid] = 1.f; Ids[tid] = input_ids[tid * 128 + t]; }
    for (int e = tid; e < 8192; e += 256) { ACC[e] = 0.f; SL[e] = 0.f; }
    __syncthreads();
    for (int e = tid; e < 4096; e += 256) {
      int b = e >> 7, d = e & 127;
      int id = Ids[b];
      float m = emb_mag[id * 128 + d], ph = emb_phase[id * 128 + d];
      Xh[b * 512 + d]       = (_Float16)(m * cosf(ph));
      Xh[b * 512 + 128 + d] = (_Float16)(m * sinf(ph));
    }
    __syncthreads();

    for (int n = 0; n < 12; ++n) {
      // ---- cell linear (complex) ----
      clin_gemm(Xh, 0, 128, cWr_t, cWi_t, T, T + 128);
      GFENCE();
      // ---- magnitude norm (ddof=1) + modReLU -> f16 acts ----
      {
        int b = tid >> 3, sub = tid & 7;
        float s = 0.f, s2 = 0.f;
        for (int q = 0; q < 16; ++q) {
          int d = sub * 16 + q;
          float r = T[b * 256 + d], iv = T[b * 256 + 128 + d];
          float m = sqrtf(r * r + iv * iv) + 1e-6f;
          Sc[b * 128 + d] = m;
          s += m; s2 += m * m;
        }
        Red[tid] = s; Red2[tid] = s2;
      }
      __syncthreads();
      if (tid < 32) {
        float s = 0.f, s2 = 0.f;
        for (int q = 0; q < 8; ++q) { s += Red[tid * 8 + q]; s2 += Red2[tid * 8 + q]; }
        float mean = s / 128.f;
        float var = (s2 - 128.f * mean * mean) / 127.f;
        RowMean[tid] = mean;
        RowRstd[tid] = rsqrtf(var + 1e-6f);
      }
      __syncthreads();
      for (int e = tid; e < 4096; e += 256) {
        int b = e >> 7, d = e & 127;
        float r = T[b * 256 + d], iv = T[b * 256 + 128 + d];
        float m = Sc[b * 128 + d];
        float nm = (m - RowMean[b]) * RowRstd[b] * norm_scale[d] + norm_shift[d];
        float rr = nm * r / m, ii = nm * iv / m;
        float n2 = sqrtf(rr * rr + ii * ii) + 1e-6f;
        float scl = fmaxf(n2 + modrelu_b[d], 0.f) / n2;
        Xh[b * 512 + d]       = (_Float16)(rr * scl);
        Xh[b * 512 + 128 + d] = (_Float16)(ii * scl);
      }
      __syncthreads();
      // ---- v linear then o linear ----
      clin_gemm(Xh, 0, 128, vWr_t, vWi_t, T, T + 128);
      GFENCE();
      for (int e = tid; e < 8192; e += 256)
        Xh[(e >> 8) * 512 + (e & 255)] = (_Float16)T[e];
      __syncthreads();
      clin_gemm(Xh, 0, 128, oWr_t, oWi_t, T, T + 128);   // T = pr|pi
      GFENCE();
      // ---- halt + stack-control heads ----
      if (tid < 32) {
        float h = halt_b[0], s0 = stk_b[0], s1 = stk_b[1], s2v = stk_b[2];
        for (int k = 0; k < 256; ++k) {
          float z = T[tid * 256 + k];
          h  += z * halt_W[k];
          s0 += z * stk_W[k * 3 + 0];
          s1 += z * stk_W[k * 3 + 1];
          s2v+= z * stk_W[k * 3 + 2];
        }
        RowHalt[tid] = 1.f / (1.f + expf(-h));
        float mx = fmaxf(s0, fmaxf(s1, s2v));
        float e0 = expf(s0 - mx), e1 = expf(s1 - mx), e2 = expf(s2v - mx);
        float inv = 1.f / (e0 + e1 + e2);
        RowPush[tid] = e0 * inv; RowPop[tid] = e1 * inv; RowNoop[tid] = e2 * inv;
      }
      __syncthreads();
      // ---- soft stack: pointer rolls ----
      float* PuS = Sc;          // old rolled-up pointer
      float* NpS = Sc + 2048;   // unnormalized new pointer
      for (int e = tid; e < 1536; e += 256) {
        int b = e / 48, j = e % 48;
        float pu = Ptr[b * 48 + ((j + 47) % 48)];
        float pd = Ptr[b * 48 + ((j + 1) % 48)];
        PuS[e] = pu;
        NpS[e] = RowPush[b] * pu + RowPop[b] * pd + RowNoop[b] * Ptr[b * 48 + j];
      }
      __syncthreads();
      if (tid < 32) {
        float s = 0.f;
        for (int j = 0; j < 48; ++j) s += NpS[tid * 48 + j];
        RowSum[tid] = 1.f / (s + 1e-6f);
      }
      __syncthreads();
      for (int e = tid; e < 1536; e += 256) Ptr[e] = NpS[e] * RowSum[e / 48];
      __syncthreads();
      // ---- memory write + weighted read; stage pr|pi|sr|si f16 for gate ----
      for (int e = tid; e < 8192; e += 256) {
        int b = e >> 8, c = e & 255;
        float zf = T[e];
        float pushb = RowPush[b];
        float racc = 0.f;
        for (int j = 0; j < 48; ++j) {
          float wm = pushb * PuS[b * 48 + j];
          size_t mi = (size_t)(b * 48 + j) * 256 + c;
          float m = memW[mi];
          m = wm * zf + (1.f - wm) * m;
          memW[mi] = m;
          racc += m * Ptr[b * 48 + j];
        }
        U[e] = racc;
        Xh[b * 512 + 256 + c] = (_Float16)racc;
        Xh[b * 512 + c]       = (_Float16)zf;
      }
      GFENCE();
      // ---- gate GEMM (32x512 @ 512x128) ----
      gemm_nk(Xh, 512, gateWt, 512, 128, G, 128);
      GFENCE();
      // ---- gated residual + halting accumulation ----
      bool lastd = (n == 11);
      for (int e = tid; e < 4096; e += 256) {
        int b = e >> 7, d = e & 127;
        float g = 1.f / (1.f + expf(-(G[b * 128 + d] + gate_b[d])));
        float pr = T[b * 256 + d],  pi = T[b * 256 + 128 + d];
        float sr = U[b * 256 + d],  si = U[b * 256 + 128 + d];
        float zr = (1.f - g) * pr + g * pi + g * (sr - si);
        float zi = (1.f - g) * pi - g * pr + g * (si + sr);
        float w = lastd ? RowRem[b] : RowRem[b] * RowHalt[b];
        ACC[b * 256 + d]       += w * zr;
        ACC[b * 256 + 128 + d] += w * zi;
        SL[b * 256 + d]        += zr * (1.f / 12.f);
        SL[b * 256 + 128 + d]  += zi * (1.f / 12.f);
        Xh[b * 512 + d]        = (_Float16)zr;
        Xh[b * 512 + 128 + d]  = (_Float16)zi;
      }
      GFENCE();
      if (tid < 32) RowRem[tid] *= (1.f - RowHalt[tid]);
      __syncthreads();
    }

    // ---- VQ (syntactic on ACC, semantic on SL) ----
    vq_step(ACC, cb_syn, ctxsyn_W, ctxsyn_b, adj_syn, 64,
            PrevSyn, NewSyn, Sc, Fx, Red, RowZn, &LossAcc);
    vq_step(SL, cb_sem, ctxsem_W, ctxsem_b, adj_sem, 128,
            PrevSem, NewSem, Sc, Fx, Red, RowZn, &LossAcc);
    // straight-through: decoder input = 0.5*(cb_syn[isyn] + cb_sem[isem])
    for (int e = tid; e < 8192; e += 256) {
      int b = e >> 8, c = e & 255;
      float x = 0.5f * (cb_syn[NewSyn[b] * 256 + c] + cb_sem[NewSem[b] * 256 + c]);
      zq16[(size_t)(t * 32 + b) * 256 + c] = (_Float16)x;
    }
    __syncthreads();
    if (tid < 32) { PrevSyn[tid] = NewSyn[tid]; PrevSem[tid] = NewSem[tid]; }
    GFENCE();
  }
  if (tid == 0) lossOut[0] = LossAcc;
}

// ---------------------------------------------------------------------------
// Phase 3: decoder GEMM (S*B=4096 x 256) @ (256 x 16384) + bias.
// Block = 32 rows x 128 cols; 8 waves, 2 m-tiles each; store-bound epilogue.
// ---------------------------------------------------------------------------
__global__ __launch_bounds__(256, 2) void crsn_decoder(
    const _Float16* __restrict__ zq16, const _Float16* __restrict__ decWt,
    const float* __restrict__ dec_b, float* __restrict__ out) {
  __shared__ __align__(16) _Float16 Ah[32 * 256];
  int tid = threadIdx.x, lane = tid & 31, wave = tid >> 5;
  int rb = blockIdx.x * 32;
  int vb = blockIdx.y * 128;

  const v8h* src = (const v8h*)(zq16 + (size_t)rb * 256);
  v8h* dst = (v8h*)Ah;
  for (int i = tid; i < 1024; i += 256) dst[i] = src[i];
  __syncthreads();

  int colBase = vb + wave * 16;
  __builtin_prefetch(decWt + (size_t)colBase * 256, 0, 1);  // global_prefetch_b8
  for (int mt = 0; mt < 2; ++mt) {
    v8f acc = {};
    for (int kb = 0; kb < 8; ++kb) {
      v16h A = load_afrag(Ah, 256, 0, mt, kb, lane);
      v16h B = load_bfrag(decWt, 256, colBase, kb, lane);
      acc = wmma16(A, B, acc);
    }
    int col = colBase + (lane & 15);
    int rbase = rb + mt * 16 + ((lane >> 4) ? 8 : 0);
    float bias = dec_b[col];
    for (int r = 0; r < 8; ++r) {
      int row = rbase + r;              // row = t*32 + b
      int tt = row >> 5, bb = row & 31;
      out[(size_t)(bb * 128 + tt) * 16384 + col] = acc[r] + bias;
    }
  }
}

// ---------------------------------------------------------------------------
extern "C" void kernel_launch(void* const* d_in, const int* in_sizes, int n_in,
                              void* d_out, int out_size, void* d_ws, size_t ws_size,
                              hipStream_t stream) {
  (void)in_sizes; (void)n_in; (void)out_size; (void)ws_size;
  const int*   input_ids  = (const int*)d_in[0];
  const int*   prev_syn   = (const int*)d_in[1];
  const int*   prev_sem   = (const int*)d_in[2];
  const float* emb_mag    = (const float*)d_in[3];
  const float* emb_phase  = (const float*)d_in[4];
  const float* cell_Wr    = (const float*)d_in[5];
  const float* cell_Wi    = (const float*)d_in[6];
  const float* norm_scale = (const float*)d_in[7];
  const float* norm_shift = (const float*)d_in[8];
  const float* modrelu_b  = (const float*)d_in[9];
  // d_in[10..13] = qW/kW : unused in the forward pass
  const float* vWr        = (const float*)d_in[14];
  const float* vWi        = (const float*)d_in[15];
  const float* oWr        = (const float*)d_in[16];
  const float* oWi        = (const float*)d_in[17];
  const float* halt_W     = (const float*)d_in[18];
  const float* halt_b     = (const float*)d_in[19];
  const float* stk_W      = (const float*)d_in[20];
  const float* stk_b      = (const float*)d_in[21];
  const float* cb_syn     = (const float*)d_in[22];
  const float* cb_sem     = (const float*)d_in[23];
  const float* ctxsyn_W   = (const float*)d_in[24];
  const float* ctxsyn_b   = (const float*)d_in[25];
  const float* ctxsem_W   = (const float*)d_in[26];
  const float* ctxsem_b   = (const float*)d_in[27];
  const float* adj_syn    = (const float*)d_in[28];
  const float* adj_sem    = (const float*)d_in[29];
  const float* gate_W     = (const float*)d_in[30];
  const float* gate_b     = (const float*)d_in[31];
  const float* dec_W      = (const float*)d_in[32];
  const float* dec_b      = (const float*)d_in[33];

  char* ws = (char*)d_ws;
  _Float16* pk     = (_Float16*)ws;
  _Float16* cWr_t  = pk;
  _Float16* cWi_t  = pk + 16384;
  _Float16* vWr_t  = pk + 32768;
  _Float16* vWi_t  = pk + 49152;
  _Float16* oWr_t  = pk + 65536;
  _Float16* oWi_t  = pk + 81920;
  _Float16* gateWt = pk + 98304;
  _Float16* decWt  = pk + 163840;
  _Float16* zq16   = pk + 4358144;
  float* fbase = (float*)(ws + 2ull * (4358144 + 1048576));
  float* memW  = fbase;
  float* T     = memW + 393216;
  float* U     = T + 8192;
  float* ACCb  = U + 8192;
  float* SLb   = ACCb + 8192;
  float* G     = SLb + 8192;

  float* out   = (float*)d_out;
  float* loss  = out + (size_t)32 * 128 * 16384;

  pack_weights<<<4096, 256, 0, stream>>>(cell_Wr, cell_Wi, vWr, vWi, oWr, oWi,
                                         gate_W, dec_W, pk);
  crsn_recurrent<<<1, 256, 0, stream>>>(
      input_ids, prev_syn, prev_sem, emb_mag, emb_phase,
      norm_scale, norm_shift, modrelu_b, halt_W, halt_b, stk_W, stk_b,
      cb_syn, cb_sem, ctxsyn_W, ctxsyn_b, ctxsem_W, ctxsem_b,
      adj_syn, adj_sem, gate_b,
      cWr_t, cWi_t, vWr_t, vWi_t, oWr_t, oWi_t, gateWt,
      zq16, memW, T, U, ACCb, SLb, G, loss);
  crsn_decoder<<<dim3(128, 128), 256, 0, stream>>>(zq16, decWt, dec_b, out);
}
